// ORGAN_59622736003980
// MI455X (gfx1250) — compile-verified, matrix operand order checked by reading, not run on hardware
//
#include <hip/hip_runtime.h>

// ---------------------------------------------------------------------------
// 2-layer LSTM autoregressive sampler for MI455X (gfx1250, wave32, WMMA).
// One workgroup (512 thr = 16 wave32) owns 16 batch rows and runs all T steps.
// GEMMs: [16 x Kcat] @ [Kcat x 2048] via v_wmma_f32_16x16x32_bf16.
// Weights pre-packed f32->bf16 into WMMA B-fragment order in d_ws (L2 resident).
// K loops are pinned to unroll(1): each iteration streams 1 A frag (LDS) +
// 4 B frags (global/L2) into 4 live accumulators -> no VGPR spills; latency
// is hidden by 16 resident waves, not by in-wave unrolling.
// ---------------------------------------------------------------------------

typedef __attribute__((ext_vector_type(16))) __bf16 v16bf;
typedef __attribute__((ext_vector_type(8)))  float  v8f;
typedef unsigned short ush;
typedef unsigned int   uint32;

#define PAD_TOK 0
#define EOS_TOK 2
#define HID   512
#define EMB   32
#define VOCAB 32
#define GATES 2048                 // 4*HID
#define NT0   17                   // (EMB + HID)/32 k-tiles, layer 0
#define NT1   32                   // (HID + HID)/32 k-tiles, layer 1
#define FRAG  512                  // elements per 32x16 bf16 B fragment
#define A0W   544                  // acat0 row width: [x(32) | h0(512)]

// workspace element counts
#define W0E ((size_t)128 * NT0 * FRAG)   // 1,114,112 bf16
#define W1E ((size_t)128 * NT1 * FRAG)   // 2,097,152 bf16
#define WOE ((size_t)2 * 16 * FRAG)      //    16,384 bf16

#define WMMA_BF16(A, B, C) \
    __builtin_amdgcn_wmma_f32_16x16x32_bf16(false, (A), false, (B), (short)0, (C), false, false)

__device__ __forceinline__ ush f2bf(float f) {
    uint32 u = __float_as_uint(f);
    uint32 r = (u + 0x7FFFu + ((u >> 16) & 1u)) >> 16;   // RNE
    return (ush)r;
}
__device__ __forceinline__ float sigf(float x) { return 1.0f / (1.0f + __expf(-x)); }
__device__ __forceinline__ uint32 mix32(uint32 x) {
    x ^= x >> 16; x *= 0x7feb352dU;
    x ^= x >> 15; x *= 0x846ca68bU;
    x ^= x >> 16; return x;
}

// GEMM over a K-tile range: acc[q] += A(kt) * B(tile = q*32 + tileBase, ktOff+kt)
// NOTE: kt loop must stay rolled -- unrolling hoists all B loads and spills.
__device__ __forceinline__ void gemm4(v8f acc[4], const ush* __restrict__ aptr, int nkt,
                                      const ush* __restrict__ wf, int wkt, int ktOff,
                                      int tileBase, int lane) {
#pragma clang loop unroll(disable)
    for (int kt = 0; kt < nkt; kt++) {
        v16bf A = *(const v16bf*)(aptr + kt * 32);
#pragma unroll
        for (int q = 0; q < 4; q++) {
            const size_t fo = ((size_t)((q * 32 + tileBase) * wkt + ktOff + kt) << 9) + (lane << 4);
            v16bf B = *(const v16bf*)(wf + fo);
            acc[q] = WMMA_BF16(A, B, acc[q]);
        }
    }
}

// bias + LSTM gates on one 16-col hid block; acc = {i,f,g,o}; updates c, returns h
__device__ __forceinline__ v8f gates4(const v8f acc[4], const float* __restrict__ bias,
                                      int tileBase, int lane, v8f& cst) {
    const int bc = (lane & 15);
    const float b0 = bias[(tileBase) * 16 + bc];
    const float b1 = bias[(32 + tileBase) * 16 + bc];
    const float b2 = bias[(64 + tileBase) * 16 + bc];
    const float b3 = bias[(96 + tileBase) * 16 + bc];
    v8f h;
#pragma unroll
    for (int r = 0; r < 8; r++) {
        float i_ = sigf(acc[0][r] + b0);
        float f_ = sigf(acc[1][r] + b1);
        float g_ = tanhf(acc[2][r] + b2);
        float o_ = sigf(acc[3][r] + b3);
        float c  = f_ * cst[r] + i_ * g_;
        cst[r] = c;
        h[r] = o_ * tanhf(c);
    }
    return h;
}

// ---------------- weight packing: f32 -> bf16 WMMA B fragments --------------
// fragment element e (0..511): lane = e/16, j = e%16
//   n = ntile*16 + (lane&15) ; k = ktile*32 + (lane>>4)*16 + j ; B[k][n] = W[n][k]

__global__ void ORGAN_pack_w0(const float* __restrict__ wih0,
                              const float* __restrict__ whh0,
                              ush* __restrict__ dst) {
    size_t idx = (size_t)blockIdx.x * 256 + threadIdx.x;
    if (idx >= W0E) return;
    int nt = (int)(idx / (NT0 * FRAG));
    int rem = (int)(idx % (NT0 * FRAG));
    int kt = rem >> 9, e = rem & 511;
    int lane = e >> 4, j = e & 15;
    int n = nt * 16 + (lane & 15);
    int k = kt * 32 + ((lane >> 4) << 4) + j;
    float v = (k < EMB) ? wih0[n * EMB + k] : whh0[n * HID + (k - EMB)];
    dst[idx] = f2bf(v);
}

__global__ void ORGAN_pack_w1(const float* __restrict__ wih1,
                              const float* __restrict__ whh1,
                              ush* __restrict__ dst) {
    size_t idx = (size_t)blockIdx.x * 256 + threadIdx.x;
    if (idx >= W1E) return;
    int nt = (int)(idx / (NT1 * FRAG));
    int rem = (int)(idx % (NT1 * FRAG));
    int kt = rem >> 9, e = rem & 511;
    int lane = e >> 4, j = e & 15;
    int n = nt * 16 + (lane & 15);
    int k = kt * 32 + ((lane >> 4) << 4) + j;
    float v = (k < HID) ? wih1[n * HID + k] : whh1[n * HID + (k - HID)];
    dst[idx] = f2bf(v);
}

__global__ void ORGAN_pack_wo(const float* __restrict__ wout,
                              ush* __restrict__ dst) {
    size_t idx = (size_t)blockIdx.x * 256 + threadIdx.x;
    if (idx >= WOE) return;
    int nt = (int)(idx / (16 * FRAG));
    int rem = (int)(idx % (16 * FRAG));
    int kt = rem >> 9, e = rem & 511;
    int lane = e >> 4, j = e & 15;
    int n = nt * 16 + (lane & 15);
    int k = kt * 32 + ((lane >> 4) << 4) + j;
    dst[idx] = f2bf(wout[n * HID + k]);
}

__global__ void ORGAN_pack_bias(const float* __restrict__ bih0, const float* __restrict__ bhh0,
                                const float* __restrict__ bih1, const float* __restrict__ bhh1,
                                const float* __restrict__ bo,
                                float* __restrict__ bias0, float* __restrict__ bias1,
                                float* __restrict__ bout) {
    int i = blockIdx.x * 256 + threadIdx.x;
    if (i < GATES)      bias0[i] = bih0[i] + bhh0[i];
    if (i < GATES)      bias1[i] = bih1[i] + bhh1[i];
    if (i < VOCAB)      bout[i]  = bo[i];
}

// -------------------------------- main kernel -------------------------------

__global__ __launch_bounds__(512)
void ORGAN_lstm_sample(const int* __restrict__ prevs, const float* __restrict__ emb,
                       const ush* __restrict__ w0f, const ush* __restrict__ w1f,
                       const ush* __restrict__ wof,
                       const float* __restrict__ bias0, const float* __restrict__ bias1,
                       const float* __restrict__ bout,
                       int* __restrict__ samples, int* __restrict__ lengths, int T) {
    __shared__ __attribute__((aligned(32))) ush acat0[16 * A0W];    // [x | h0], 17 KB
    __shared__ __attribute__((aligned(32))) ush h1buf[16 * HID];    // 16 KB
    __shared__ float logitsS[16 * VOCAB];
    __shared__ int tokS[16], endS[16], lenS[16];

    const int tid  = threadIdx.x;
    const int wave = tid >> 5;
    const int lane = tid & 31;
    const int rowBase = blockIdx.x * 16;
    const int arow  = lane & 15;   // A-fragment row (M)
    const int khalf = lane >> 4;   // A-fragment K half
    const int tb0 = 2 * wave;      // tile base for hid block j=0; j=1 -> tb0+1

    // ---- init ----
    for (int i = tid; i < 16 * A0W; i += 512) acat0[i] = 0;
    for (int i = tid; i < 16 * HID; i += 512) h1buf[i] = 0;
    if (tid < 16) {
        int tk = prevs[rowBase + tid];
        tokS[tid] = tk;
        endS[tid] = (tk == EOS_TOK) ? 1 : 0;
        lenS[tid] = 0;
        samples[(size_t)(rowBase + tid) * (T + 1)] = tk;
    }
    const v8f vzero = {0.f, 0.f, 0.f, 0.f, 0.f, 0.f, 0.f, 0.f};
    v8f c0a = vzero, c0b = vzero, c1a = vzero, c1b = vzero;   // cell state, 2 hid blocks/layer
    __syncthreads();

    const ush* a0p = &acat0[arow * A0W + khalf * 16];         // layer0 A base (x|h0)
    const ush* a1p = &acat0[arow * A0W + 32 + khalf * 16];    // h0 region
    const ush* a2p = &h1buf[arow * HID + khalf * 16];         // h1 region
    const int hrow = khalf << 3;                              // C-tile M offset
    const int hcol = (wave << 5) + (lane & 15);               // hid col, block j adds j*16

    for (int t = 0; t < T; t++) {
        // ---- embedding gather: x = emb[tok], bf16 (512 threads = 16x32) ----
        {
            int r = tid >> 5, c = tid & 31;
            acat0[r * A0W + c] = f2bf(emb[tokS[r] * EMB + c]);
        }
        __syncthreads();

        // ---- layer 0: g = [x | h0] @ [W_ih0 | W_hh0]^T  (K = 544) ----
        v8f h0ra, h0rb;
        {
            v8f acc[4] = {vzero, vzero, vzero, vzero};
            gemm4(acc, a0p, NT0, w0f, NT0, 0, tb0, lane);
            h0ra = gates4(acc, bias0, tb0, lane, c0a);
        }
        {
            v8f acc[4] = {vzero, vzero, vzero, vzero};
            gemm4(acc, a0p, NT0, w0f, NT0, 0, tb0 + 1, lane);
            h0rb = gates4(acc, bias0, tb0 + 1, lane, c0b);
        }
        __syncthreads();   // all waves finished reading h0 region
#pragma unroll
        for (int r = 0; r < 8; r++) {
            acat0[(hrow + r) * A0W + 32 + hcol]      = f2bf(h0ra[r]);
            acat0[(hrow + r) * A0W + 32 + hcol + 16] = f2bf(h0rb[r]);
        }
        __syncthreads();

        // ---- layer 1: g = [h0 | h1] @ [W_ih1 | W_hh1]^T  (K = 1024) ----
        v8f h1ra, h1rb;
        {
            v8f acc[4] = {vzero, vzero, vzero, vzero};
            gemm4(acc, a1p, 16, w1f, NT1, 0,  tb0, lane);   // h0 half
            gemm4(acc, a2p, 16, w1f, NT1, 16, tb0, lane);   // h1 half
            h1ra = gates4(acc, bias1, tb0, lane, c1a);
        }
        {
            v8f acc[4] = {vzero, vzero, vzero, vzero};
            gemm4(acc, a1p, 16, w1f, NT1, 0,  tb0 + 1, lane);
            gemm4(acc, a2p, 16, w1f, NT1, 16, tb0 + 1, lane);
            h1rb = gates4(acc, bias1, tb0 + 1, lane, c1b);
        }
        __syncthreads();   // all waves finished reading h1buf
#pragma unroll
        for (int r = 0; r < 8; r++) {
            h1buf[(hrow + r) * HID + hcol]      = f2bf(h1ra[r]);
            h1buf[(hrow + r) * HID + hcol + 16] = f2bf(h1rb[r]);
        }
        __syncthreads();

        // ---- logits: [16 x 512] @ W_out^T -> [16 x 32] (waves 0,1) ----
        if (wave < 2) {
            v8f lacc = vzero;
#pragma clang loop unroll(disable)
            for (int kt = 0; kt < 16; kt++) {
                v16bf A = *(const v16bf*)(a2p + kt * 32);
                v16bf B = *(const v16bf*)(wof + ((size_t)(wave * 16 + kt) << 9) + (lane << 4));
                lacc = WMMA_BF16(A, B, lacc);
            }
            float bv = bout[wave * 16 + (lane & 15)];
#pragma unroll
            for (int r = 0; r < 8; r++)
                logitsS[(hrow + r) * VOCAB + wave * 16 + (lane & 15)] = lacc[r] + bv;
        }
        __syncthreads();

        // ---- Gumbel-max categorical sampling (wave 0, lane == vocab id) ----
        if (wave == 0) {
            for (int row = 0; row < 16; row++) {
                float logit = logitsS[row * VOCAB + lane];
                uint32 h = mix32(((uint32)t * 0x9E3779B9u) ^
                                 mix32((uint32)(rowBase + row) * 0x85EBCA6Bu +
                                       (uint32)lane * 0xC2B2AE35u + 0x27220A95u));
                float u = ((float)(h >> 8) + 0.5f) * (1.0f / 16777216.0f);
                float key = logit - __logf(-__logf(u));
                int best = lane;
#pragma unroll
                for (int m = 16; m >= 1; m >>= 1) {
                    float ok = __shfl_xor(key, m, 32);
                    int   ob = __shfl_xor(best, m, 32);
                    if (ok > key || (ok == key && ob < best)) { key = ok; best = ob; }
                }
                if (lane == 0) {
                    int e   = endS[row];
                    int cur = e ? PAD_TOK : best;
                    lenS[row] += e ? 0 : 1;
                    endS[row]  = e | (cur == EOS_TOK);
                    tokS[row]  = cur;
                    samples[(size_t)(rowBase + row) * (T + 1) + t + 1] = cur;
                }
            }
        }
        __syncthreads();
    }

    if (tid < 16) lengths[rowBase + tid] = lenS[tid] + 1;
}

// -------------------------------- launcher ----------------------------------

extern "C" void kernel_launch(void* const* d_in, const int* in_sizes, int n_in,
                              void* d_out, int out_size, void* d_ws, size_t ws_size,
                              hipStream_t stream) {
    const int*   prevs = (const int*)  d_in[0];
    const float* emb   = (const float*)d_in[1];
    const float* Wih0  = (const float*)d_in[2];
    const float* Whh0  = (const float*)d_in[3];
    const float* bih0  = (const float*)d_in[4];
    const float* bhh0  = (const float*)d_in[5];
    const float* Wih1  = (const float*)d_in[6];
    const float* Whh1  = (const float*)d_in[7];
    const float* bih1  = (const float*)d_in[8];
    const float* bhh1  = (const float*)d_in[9];
    const float* Wout  = (const float*)d_in[10];
    const float* bo    = (const float*)d_in[11];
    (void)n_in; (void)ws_size;

    const int N = in_sizes[0];                 // 4096 rows
    const int Tp1 = out_size / N - 1;          // samples is N x (T+1), plus N lengths
    const int T = Tp1 - 1;

    // workspace carve-up (bf16 weight fragments + fused biases)
    ush* w0f = (ush*)d_ws;
    ush* w1f = w0f + W0E;
    ush* wof = w1f + W1E;
    float* bias0 = (float*)(wof + WOE);
    float* bias1 = bias0 + GATES;
    float* boutp = bias1 + GATES;

    ORGAN_pack_w0<<<(unsigned)((W0E + 255) / 256), 256, 0, stream>>>(Wih0, Whh0, w0f);
    ORGAN_pack_w1<<<(unsigned)((W1E + 255) / 256), 256, 0, stream>>>(Wih1, Whh1, w1f);
    ORGAN_pack_wo<<<(unsigned)((WOE + 255) / 256), 256, 0, stream>>>(Wout, wof);
    ORGAN_pack_bias<<<(GATES + 255) / 256, 256, 0, stream>>>(bih0, bhh0, bih1, bhh1, bo,
                                                            bias0, bias1, boutp);

    int* samples = (int*)d_out;
    int* lengths = samples + (size_t)N * Tp1;
    ORGAN_lstm_sample<<<N / 16, 512, 0, stream>>>(prevs, emb, w0f, w1f, wof,
                                                  bias0, bias1, boutp,
                                                  samples, lengths, T);
}
